// SynthesisLayer_29111288333087
// MI455X (gfx1250) — compile-verified, matrix operand order checked by reading, not run on hardware
//
#include <hip/hip_runtime.h>
#include <hip/hip_bf16.h>

// ---------------------------------------------------------------------------
// StyleGAN3 synthesis layer for MI455X (gfx1250, wave32, WMMA + async-to-LDS).
//   k_styles : styles = norm(w @ aw.T/sqrt(512) + ab)
//   k_xmod   : padded channels-last bf16 xmod[b][68][68][512] = x * s (halo=2)
//   k_wprep  : per-cout RMS-norm weights -> bf16 W[t][o][i] + q[o][i]
//   k_demod  : d[b,o] = rsqrt(sum_i q[o,i] s[b,i]^2 + eps)
//   k_conv   : implicit GEMM on v_wmma_f32_16x16x32_bf16, double-buffered LDS
//              tiles filled with global_load_async_to_lds_b128
//   k_fir    : fused up2-FIR12 -> leakyReLU*sqrt2*clamp -> FIR12-down2 in LDS
// ---------------------------------------------------------------------------

typedef __attribute__((ext_vector_type(16))) __bf16 v16bf;
typedef __attribute__((ext_vector_type(8)))  float  v8f;
typedef __attribute__((ext_vector_type(4)))  int    v4i;

#define CIN   512
#define COUT  512
#define BATCH 8
#define YW    66
#define NPOS  (YW*YW)          // 4356
#define KTAPS 9
#define PW    68               // padded width (halo 2 each side)

union FragBF { unsigned u[8]; v16bf v; };
union PackBF { __bf16 h[2]; unsigned u; };

typedef __attribute__((address_space(3))) __bf16 lds_bf16;
typedef __attribute__((address_space(1))) v4i*   gv4i_p;   // global v4i*
typedef __attribute__((address_space(3))) v4i*   lv4i_p;   // LDS v4i*

// 16-byte global -> LDS async copy (gfx1250, tracked by ASYNCcnt).
__device__ __forceinline__ void cp_async16(__bf16* lds, const __bf16* g) {
#if __has_builtin(__builtin_amdgcn_global_load_async_to_lds_b128)
    __builtin_amdgcn_global_load_async_to_lds_b128(
        (gv4i_p)(const v4i*)g, (lv4i_p)(v4i*)lds, 0, 0);
#else
    const unsigned ldsOff =
        (unsigned)(unsigned long long)(lds_bf16*)lds;   // raw LDS byte offset
    asm volatile("global_load_async_to_lds_b128 %0, %1, off"
                 :: "v"(ldsOff),
                    "v"((unsigned long long)reinterpret_cast<uintptr_t>(g)),
                    "v"(lds)                            // escape the LDS object
                 : "memory");
#endif
}

__device__ __forceinline__ void wait_async2() {
#if __has_builtin(__builtin_amdgcn_s_wait_asynccnt)
    __builtin_amdgcn_s_wait_asynccnt(2);
#else
    asm volatile("s_wait_asynccnt 2" ::: "memory");
#endif
}
__device__ __forceinline__ void wait_async0() {
#if __has_builtin(__builtin_amdgcn_s_wait_asynccnt)
    __builtin_amdgcn_s_wait_asynccnt(0);
#else
    asm volatile("s_wait_asynccnt 0" ::: "memory");
#endif
}

// ---------------- styles: s = norm(w @ aw.T/sqrt(512) + ab) -----------------
__global__ __launch_bounds__(256) void k_styles(const float* __restrict__ w,
                                                const float* __restrict__ aw,
                                                const float* __restrict__ ab,
                                                float* __restrict__ s_out) {
    __shared__ float sl[BATCH * CIN];
    __shared__ float red[256];
    const int tid = threadIdx.x;
    const float inv = 0.044194173824159216f;  // 1/sqrt(512)
    for (int e = tid; e < BATCH * CIN; e += 256) {
        const int b = e >> 9, i = e & 511;
        const float* wr = w  + b * CIN;
        const float* ar = aw + i * CIN;
        float acc = 0.f;
        for (int k = 0; k < CIN; ++k) acc += wr[k] * ar[k];
        sl[e] = acc * inv + ab[i];
    }
    __syncthreads();
    float ss = 0.f;
    for (int e = tid; e < BATCH * CIN; e += 256) ss += sl[e] * sl[e];
    red[tid] = ss;
    __syncthreads();
    for (int o = 128; o > 0; o >>= 1) {
        if (tid < o) red[tid] += red[tid + o];
        __syncthreads();
    }
    const float scale = rsqrtf(red[0] / (float)(BATCH * CIN));
    for (int e = tid; e < BATCH * CIN; e += 256) s_out[e] = sl[e] * scale;
}

// ---- xmod: padded channels-last pre-modulated bf16, via LDS transpose ------
__global__ __launch_bounds__(256) void k_xmod(const float* __restrict__ x,
                                              const float* __restrict__ s,
                                              __bf16* __restrict__ xm) {
    __shared__ float tile[64][65];
    const int py = blockIdx.x;     // padded row 0..67
    const int b  = blockIdx.y;
    const int tid = threadIdx.x;
    __bf16* base = xm + ((size_t)(b * PW) + py) * PW * CIN;
    const int iy = py - 2;
    if (iy < 0 || iy >= 64) {       // top/bottom halo rows: all zero
        for (int e = tid; e < PW * CIN; e += 256) base[e] = (__bf16)0.f;
        return;
    }
    for (int e = tid; e < 2 * CIN; e += 256) {   // left/right halo columns
        base[e]            = (__bf16)0.f;
        base[66 * CIN + e] = (__bf16)0.f;
    }
    const int lx = tid & 63, li = tid >> 6;        // load: lanes along x
    const int cl = (tid & 31) * 2, pxg = tid >> 5; // store: lanes along chan
    for (int ic = 0; ic < CIN; ic += 64) {
        __syncthreads();
        #pragma unroll
        for (int r = 0; r < 16; ++r) {
            const int i = li + r * 4;
            tile[i][lx] = x[((size_t)(b * CIN + ic + i) * 64 + iy) * 64 + lx];
        }
        __syncthreads();
        const float s0 = s[b * CIN + ic + cl];
        const float s1 = s[b * CIN + ic + cl + 1];
        #pragma unroll
        for (int r = 0; r < 8; ++r) {
            const int px = pxg + r * 8;
            PackBF pk;
            pk.h[0] = (__bf16)(tile[cl][px] * s0);
            pk.h[1] = (__bf16)(tile[cl + 1][px] * s1);
            *(unsigned*)&base[(size_t)(px + 2) * CIN + ic + cl] = pk.u;
        }
    }
}

// ------- weight prep: per-o RMS norm, emit bf16 W[t][o][i] + q[o][i] --------
__global__ __launch_bounds__(256) void k_wprep(const float* __restrict__ cw,
                                               __bf16* __restrict__ wT,
                                               float* __restrict__ q) {
    __shared__ float red[256];
    const int o = blockIdx.x, tid = threadIdx.x;
    const float* wo = cw + (size_t)o * (CIN * KTAPS);
    float ss = 0.f;
    for (int e = tid; e < CIN * KTAPS; e += 256) { float v = wo[e]; ss += v * v; }
    red[tid] = ss;
    __syncthreads();
    for (int s = 128; s > 0; s >>= 1) {
        if (tid < s) red[tid] += red[tid + s];
        __syncthreads();
    }
    const float scale = rsqrtf(red[0] / (float)(CIN * KTAPS));
    for (int i = tid; i < CIN; i += 256) {
        float qs = 0.f;
        #pragma unroll
        for (int t = 0; t < KTAPS; ++t) {
            float v = wo[i * KTAPS + t] * scale;
            qs += v * v;
            wT[(size_t)t * (COUT * CIN) + (size_t)o * CIN + i] = (__bf16)v;
        }
        q[o * CIN + i] = qs;
    }
}

// ------------- demod: d[b,o] = rsqrt(sum_i q[o,i]*s[b,i]^2 + eps) -----------
__global__ __launch_bounds__(512) void k_demod(const float* __restrict__ q,
                                               const float* __restrict__ s,
                                               float* __restrict__ dmod) {
    __shared__ float s2[CIN];
    const int b = blockIdx.x, tid = threadIdx.x;
    const float sv = s[b * CIN + tid];
    s2[tid] = sv * sv;
    __syncthreads();
    float acc = 1e-8f;
    const float* qr = q + (size_t)tid * CIN;
    for (int i = 0; i < CIN; ++i) acc += qr[i] * s2[i];
    dmod[b * COUT + tid] = rsqrtf(acc);
}

// --------------------- WMMA implicit-GEMM modulated conv --------------------
// grid = (69, 8, 8), block 256 (8 wave32).
// WG tile: 64 cout x 64 positions; wave -> 16x32 via two 16x16x32 accums.
#define ROWS   40                      // row stride in bf16 (32 data + 8 pad)
#define NIT    (KTAPS * (CIN / 32))    // 144

__global__ __launch_bounds__(256) void k_conv(const __bf16* __restrict__ wT,
                                              const __bf16* __restrict__ xm,
                                              const float* __restrict__ dmod,
                                              const float* __restrict__ cb,
                                              float* __restrict__ y) {
    __shared__ __bf16 SM[2][2][64][ROWS];   // [buf][A|B][row][k]

    const int tid = threadIdx.x;
    const int b  = blockIdx.z;
    const int Mt = blockIdx.y * 64;
    const int Nt = blockIdx.x * 64;

    const int lane = tid & 31, wid = tid >> 5;
    const int wm = wid & 3, wn = wid >> 2;
    const int l16 = lane & 15, half = lane >> 4;

    // staging coords: row = tid/4, 16B chunk = tid%4  (A and B identical)
    const int row = tid >> 2;
    const int ch8 = (tid & 3) * 8;

    // B global row base: per-row spatial position (clamped; junk cols unused)
    const int pos = Nt + row;
    const int pc  = pos < NPOS ? pos : NPOS - 1;
    const int yy = pc / YW, xx = pc % YW;
    const __bf16* pB = xm + ((size_t)(b * PW + yy) * PW + xx) * CIN + ch8;
    const __bf16* pA = wT + (size_t)(Mt + row) * CIN + ch8;

    v8f acc0 = {};
    v8f acc1 = {};

    const int mrow = wm * 16 + l16;
    const int n0   = wn * 32 + l16;

    auto issue = [&](int it, int buf) {
        const int t  = it >> 4;
        const int kc = (it & 15) << 5;
        cp_async16(&SM[buf][0][row][ch8], pA + (size_t)t * (COUT * CIN) + kc);
        cp_async16(&SM[buf][1][row][ch8],
                   pB + (size_t)((t / 3) * PW + (t % 3)) * CIN + kc);
    };

    issue(0, 0);
    int cur = 0;
    for (int it = 0; it < NIT; ++it) {
        if (it + 1 < NIT) {
            issue(it + 1, cur ^ 1);
            wait_async2();             // own cur-tile loads complete
        } else {
            wait_async0();
        }
        __syncthreads();               // all waves' cur tile visible

        FragBF fa, fb0, fb1;
        #pragma unroll
        for (int v = 0; v < 8; ++v) {
            const int k0 = (v < 4) ? (8 * half + 2 * v)
                                   : (16 + 8 * half + 2 * (v - 4));
            fa.u[v]  = *(const unsigned*)&SM[cur][0][mrow][k0];
            fb0.u[v] = *(const unsigned*)&SM[cur][1][n0][k0];
            fb1.u[v] = *(const unsigned*)&SM[cur][1][n0 + 16][k0];
        }
        acc0 = __builtin_amdgcn_wmma_f32_16x16x32_bf16(
            false, fa.v, false, fb0.v, (short)0, acc0, false, false);
        acc1 = __builtin_amdgcn_wmma_f32_16x16x32_bf16(
            false, fa.v, false, fb1.v, (short)0, acc1, false, false);

        __syncthreads();               // compute done before buf^1 overwrite
        cur ^= 1;
    }

    // ---- epilogue: demodulate + bias, scatter to y (b, o, 66, 66)
    const int mbase = Mt + wm * 16 + half * 8;
    const size_t ybase = (size_t)b * COUT * NPOS;
    #pragma unroll
    for (int sub = 0; sub < 2; ++sub) {
        const v8f acc = sub ? acc1 : acc0;
        const int ng = Nt + wn * 32 + sub * 16 + l16;
        if (ng < NPOS) {
            #pragma unroll
            for (int r = 0; r < 8; ++r) {
                const int o = mbase + r;
                y[ybase + (size_t)o * NPOS + ng] =
                    acc[r] * dmod[b * COUT + o] + cb[o];
            }
        }
    }
}

// ----------------------- fused up-FIR / act / down-FIR ----------------------
__device__ double bessel_i0(double xv) {
    double s = 1.0, term = 1.0;
    for (int k = 1; k < 32; ++k) {
        const double z = xv / (2.0 * k);
        term *= z * z;
        s += term;
    }
    return s;
}

__global__ __launch_bounds__(256) void k_fir(const float* __restrict__ y,
                                             float* __restrict__ out) {
    extern __shared__ float smem[];
    float* ly = smem;                 // 4356
    float* tv = ly + NPOS;            // 138*66 = 9108
    float* dh = tv + 138 * 66;        // 138*64 = 8832
    float* f  = dh + 138 * 64;        // 12 taps (fu == fd here)

    const int tid = threadIdx.x;
    const size_t chn = blockIdx.x;    // b*512 + c

    if (tid == 0) {
        const double PI = 3.14159265358979323846;
        const double a = 2.285 * 11.0 * PI * (32.0 / 64.0) + 7.95;
        const double beta = 0.5842 * pow(a - 21.0, 0.4) + 0.07886 * (a - 21.0);
        const double i0b = bessel_i0(beta);
        double h[12], sum = 0.0;
        for (int n = 0; n < 12; ++n) {
            const double m = n - 5.5;
            const double arg = PI * 0.5 * m;
            const double snc = sin(arg) / arg;
            const double kx = m / 5.5;
            const double win = bessel_i0(beta * sqrt(1.0 - kx * kx)) / i0b;
            h[n] = 0.5 * snc * win;
            sum += h[n];
        }
        for (int n = 0; n < 12; ++n) f[n] = (float)(h[n] / sum);
    }

    const float* src = y + chn * NPOS;
    for (int e = tid; e < NPOS; e += 256) ly[e] = src[e];
    __syncthreads();

    for (int e = tid; e < 138 * 66; e += 256) {   // vertical up-conv
        const int p = e / 66, col = e % 66;
        float acc = 0.f;
        const int j0 = (p & 1) ? 0 : 1;
        #pragma unroll
        for (int j = 0; j < 6; ++j) {
            const int jj = j0 + 2 * j;
            const int r = p + jj;
            const int rw = (r - 9) >> 1;
            if (r >= 9 && rw < 66) acc += f[11 - jj] * ly[rw * 66 + col];
        }
        tv[e] = 4.f * acc;
    }
    __syncthreads();

    for (int e = tid; e < 138 * 64; e += 256) {   // h-up + act + h-down
        const int p = e >> 6, ox = e & 63;
        const float* tvp = tv + p * 66;
        float accd = 0.f;
        #pragma unroll
        for (int i = 0; i < 12; ++i) {
            const int qq = 2 * ox + i;
            float av = 0.f;
            const int ii0 = (qq & 1) ? 0 : 1;
            #pragma unroll
            for (int u = 0; u < 6; ++u) {
                const int ii = ii0 + 2 * u;
                const int rr = qq + ii;
                const int m = (rr - 9) >> 1;
                if (rr >= 9 && m < 66) av += f[11 - ii] * tvp[m];
            }
            av = (av >= 0.f ? av : av * 0.2f) * 1.41421356f;
            av = fminf(fmaxf(av, -256.f), 256.f);
            accd += f[11 - i] * av;
        }
        dh[e] = accd;
    }
    __syncthreads();

    float* dst = out + chn * 4096;                // v-down -> 64x64
    for (int e = tid; e < 4096; e += 256) {
        const int oy = e >> 6, ox = e & 63;
        float acc = 0.f;
        #pragma unroll
        for (int j = 0; j < 12; ++j)
            acc += f[11 - j] * dh[(2 * oy + j) * 64 + ox];
        dst[e] = acc;
    }
}

// ---------------------------------------------------------------------------
extern "C" void kernel_launch(void* const* d_in, const int* in_sizes, int n_in,
                              void* d_out, int out_size, void* d_ws, size_t ws_size,
                              hipStream_t stream) {
    (void)in_sizes; (void)n_in; (void)out_size; (void)ws_size;
    const float* x  = (const float*)d_in[0];
    const float* w  = (const float*)d_in[1];
    const float* aw = (const float*)d_in[2];
    const float* ab = (const float*)d_in[3];
    const float* cw = (const float*)d_in[4];
    const float* cb = (const float*)d_in[5];
    float* outp = (float*)d_out;

    char* ws = (char*)d_ws;
    const size_t OFF_S  = 0;
    const size_t OFF_D  = 16 * 1024;
    const size_t OFF_Q  = 32 * 1024;
    const size_t OFF_WT = OFF_Q + (size_t)COUT * CIN * 4;            // q: 1 MB
    const size_t OFF_Y  = OFF_WT + (size_t)KTAPS * COUT * CIN * 2;   // wT: 4.5 MB
    const size_t OFF_XM = OFF_Y + (size_t)BATCH * COUT * NPOS * 4;   // y: 71.4 MB
    float*  s_ws  = (float*)(ws + OFF_S);
    float*  d_dm  = (float*)(ws + OFF_D);
    float*  q_ws  = (float*)(ws + OFF_Q);
    __bf16* wT_ws = (__bf16*)(ws + OFF_WT);
    float*  y_ws  = (float*)(ws + OFF_Y);
    __bf16* xm_ws = (__bf16*)(ws + OFF_XM);                          // 37.9 MB

    k_styles<<<dim3(1),         dim3(256), 0, stream>>>(w, aw, ab, s_ws);
    k_xmod  <<<dim3(PW, BATCH), dim3(256), 0, stream>>>(x, s_ws, xm_ws);
    k_wprep <<<dim3(COUT),      dim3(256), 0, stream>>>(cw, wT_ws, q_ws);
    k_demod <<<dim3(BATCH),     dim3(512), 0, stream>>>(q_ws, s_ws, d_dm);
    k_conv  <<<dim3(69, 8, BATCH), dim3(256), 0, stream>>>(wT_ws, xm_ws, d_dm, cb, y_ws);
    const int firShmem = (NPOS + 138 * 66 + 138 * 64 + 16) * 4;      // ~89 KB
    k_fir   <<<dim3(BATCH * COUT), dim3(256), firShmem, stream>>>(y_ws, outp);
}
